// LookWhereDownstream_61856118997148
// MI455X (gfx1250) — matrix operand order repaired and checked by reading, not compiled, not run on hardware
//
#include <hip/hip_runtime.h>
#include <hip/hip_bf16.h>
#include <stdint.h>

// ======================= types / helpers =======================
typedef unsigned short bf16_t;
typedef __attribute__((ext_vector_type(16))) __bf16 v16bf;
typedef __attribute__((ext_vector_type(8)))  float  v8f;

#if defined(__has_builtin)
#  if __has_builtin(__builtin_amdgcn_tensor_load_to_lds) && __has_builtin(__builtin_amdgcn_s_wait_tensorcnt)
#    define USE_TDM 1
#  endif
#endif
#ifndef USE_TDM
#define USE_TDM 0
#endif

#if USE_TDM
typedef __attribute__((ext_vector_type(4))) unsigned int v4u32;
typedef __attribute__((ext_vector_type(8))) int v8i32;
typedef __attribute__((ext_vector_type(4))) int v4i32;
#endif

__device__ __forceinline__ bf16_t f2bf(float f) {
  uint32_t u = __float_as_uint(f);
  u += 0x7fffu + ((u >> 16) & 1u);
  return (bf16_t)(u >> 16);
}
__device__ __forceinline__ float gelu_exact(float x) {
  return 0.5f * x * (1.0f + erff(x * 0.70710678118654752f));
}

union ABu { v16bf v; bf16_t s[16]; uint4 q[2]; };
union CFu { v8f  v; float  f[8]; };

#if USE_TDM
// Issue one 32-row x 64-col bf16 tile DMA (2D descriptor, row stride = lda elems).
__device__ __forceinline__ void tdm_load_tile(const bf16_t* gptr, int lda, uint32_t ldsoff) {
  uint64_t ga = (uint64_t)(uintptr_t)gptr;
  v4u32 g0;
  g0[0] = 1u;                                     // count=1, user descriptor
  g0[1] = ldsoff;                                 // lds_addr (bytes)
  g0[2] = (uint32_t)ga;                           // global_addr[31:0]
  g0[3] = (uint32_t)((ga >> 32) & 0x01FFFFFFu) | (2u << 30); // addr[56:32], type=2
  v8i32 g1;
  g1[0] = (int)(1u << 16);                        // wg_mask=0, data_size=2B
  g1[1] = (int)(((uint32_t)lda & 0xFFFFu) << 16); // tensor_dim0 lo16
  g1[2] = (int)(((uint32_t)lda >> 16) & 0xFFFFu); // tensor_dim0 hi16, tensor_dim1 lo16=0
  g1[3] = (int)(0x40u | (64u << 16));             // tensor_dim1=0x400000, tile_dim0=64
  g1[4] = 32;                                     // tile_dim1=32, tile_dim2=0
  g1[5] = lda;                                    // tensor_dim0_stride lo32
  g1[6] = 0;
  g1[7] = 0;
  v4i32 z4 = {0, 0, 0, 0};
  v8i32 z8 = {0, 0, 0, 0, 0, 0, 0, 0};
  __builtin_amdgcn_tensor_load_to_lds(g0, g1, z4, z4, z8, 0);
}
#endif

// ======================= GEMM: bf16 activations x packed bf16 weights =======================
// A: row-major bf16 [M x lda] (M % 32 == 0, lda % 64 == 0 region), Wt: K-major bf16
// (Wt[n*ldw+k] = W[k][n]). Block = 32 rows x (8 waves * 16 cols). Double-buffered TDM
// staging of 32x64 A tiles; 4 WMMAs per wave per stage (2 M-subtiles x 2 K-substeps).
__global__ __launch_bounds__(256) void k_gemm_wp(
    const bf16_t* __restrict__ A, int lda,
    const bf16_t* __restrict__ Wt, int ldw,
    const float* __restrict__ bias,
    float* __restrict__ outF, bf16_t* __restrict__ outB, int ldc,
    int Np, int Kp, int act)
{
  __shared__ __align__(16) bf16_t As[2][32 * 64];
  const int wave = threadIdx.x >> 5;
  const int lane = threadIdx.x & 31;
  const int m0   = blockIdx.x * 32;
  const int nt   = blockIdx.y * 8 + wave;
  const bool active = (nt * 16) < Np;
  const int col  = nt * 16 + (lane & 15);
  const int kh   = lane >> 4;
  const int arow = lane & 15;

  CFu c0, c1;
  #pragma unroll
  for (int i = 0; i < 8; ++i) { c0.f[i] = 0.f; c1.f[i] = 0.f; }

#if USE_TDM
  if (wave == 0)
    tdm_load_tile(A + (size_t)m0 * lda, lda, (uint32_t)(uintptr_t)(&As[0][0]));
#endif
  int parity = 0;
  for (int kb = 0; kb < Kp; kb += 64, parity ^= 1) {
#if USE_TDM
    if (wave == 0) {
      const bool more = (kb + 64) < Kp;
      if (more)
        tdm_load_tile(A + (size_t)m0 * lda + kb + 64, lda,
                      (uint32_t)(uintptr_t)(&As[parity ^ 1][0]));
      // TDM ops from one wave complete in order: <=1 outstanding => current tile landed.
      if (more) __builtin_amdgcn_s_wait_tensorcnt(1);
      else      __builtin_amdgcn_s_wait_tensorcnt(0);
    }
    const bf16_t* tile = &As[parity][0];
#else
    {
      int t = threadIdx.x;
      int r = t >> 3, cp = (t & 7) * 8;           // 256 threads * 16B = 32x64 bf16
      *(uint4*)&As[0][r * 64 + cp] =
          *(const uint4*)&A[(size_t)(m0 + r) * lda + kb + cp];
    }
    const bf16_t* tile = &As[0][0];
#endif
    __syncthreads();                               // tile ready for all waves
    if (active) {
      const bf16_t* wp = &Wt[(size_t)col * ldw + kb];
      #pragma unroll
      for (int ks = 0; ks < 2; ++ks) {
        ABu b;
        b.q[0] = *(const uint4*)&wp[ks * 32 + kh * 8];      // K = ks*32 + kh*8 .., then +16
        b.q[1] = *(const uint4*)&wp[ks * 32 + 16 + kh * 8];
        ABu a0, a1;
        const bf16_t* r0 = &tile[arow * 64 + ks * 32];
        const bf16_t* r1 = &tile[(16 + arow) * 64 + ks * 32];
        a0.q[0] = *(const uint4*)&r0[kh * 8];
        a0.q[1] = *(const uint4*)&r0[16 + kh * 8];
        a1.q[0] = *(const uint4*)&r1[kh * 8];
        a1.q[1] = *(const uint4*)&r1[16 + kh * 8];
        c0.v = __builtin_amdgcn_wmma_f32_16x16x32_bf16(false, a0.v, false, b.v,
                                                       (short)0, c0.v, false, false);
        c1.v = __builtin_amdgcn_wmma_f32_16x16x32_bf16(false, a1.v, false, b.v,
                                                       (short)0, c1.v, false, false);
      }
      if (kb + 64 < Kp) __builtin_prefetch(wp + 64, 0, 0);
    }
    __syncthreads();                               // all reads done before overwrite
  }
  if (!active) return;
  float bs = bias ? bias[col] : 0.f;
  const int r0 = m0 + (kh << 3);                   // lanes 16..31 -> M += 8
  #pragma unroll
  for (int v = 0; v < 8; ++v) {
    float x = c0.f[v] + bs;
    float y = c1.f[v] + bs;
    if (act == 1) { x = gelu_exact(x); y = gelu_exact(y); }
    size_t o0 = (size_t)(r0 + v) * ldc + col;
    size_t o1 = (size_t)(r0 + 16 + v) * ldc + col;
    if (outF) { outF[o0] = x; outF[o1] = y; }
    if (outB) { outB[o0] = f2bf(x); outB[o1] = f2bf(y); }
  }
}

// ======================= batched GEMM, f32 inputs converted on the fly =======================
// A: f32 [.. x lda]; Bt: f32 transposed-B, Bt[n*ldb+k]; batch bz -> (b = bz/nh, h = bz%nh).
__global__ __launch_bounds__(256) void k_bgemm_bt(
    const float* __restrict__ A, long long sAb, long long sAh, int lda,
    const float* __restrict__ Bt, long long sBb, long long sBh, int ldb,
    float* __restrict__ outF, bf16_t* __restrict__ outB,
    long long sCb, long long sCh, int ldc,
    int Ncols, int Kdim, int nh, float ascale)
{
  __shared__ __align__(16) bf16_t As[16 * 32];
  const int bz = blockIdx.z;
  const int b = bz / nh, h = bz % nh;
  const float* Ab = A  + (long long)b * sAb + (long long)h * sAh;
  const float* Bb = Bt + (long long)b * sBb + (long long)h * sBh;
  const long long cb = (long long)b * sCb + (long long)h * sCh;

  const int wave = threadIdx.x >> 5;
  const int lane = threadIdx.x & 31;
  const int m0   = blockIdx.x * 16;
  const int nt   = blockIdx.y * 8 + wave;
  const bool active = (nt * 16) < Ncols;
  const int col = nt * 16 + (lane & 15);
  const int kh  = lane >> 4;

  CFu c;
  #pragma unroll
  for (int i = 0; i < 8; ++i) c.f[i] = 0.f;

  for (int kb = 0; kb < Kdim; kb += 32) {
    __syncthreads();
    for (int i = threadIdx.x; i < 512; i += 256) {
      int r = i >> 5, cc = i & 31;
      As[r * 32 + cc] = f2bf(Ab[(size_t)(m0 + r) * lda + kb + cc] * ascale);
    }
    __syncthreads();
    if (active) {
      ABu a, bfr;
      const bf16_t* ar = &As[(lane & 15) * 32];
      a.q[0] = *(const uint4*)&ar[kh * 16];
      a.q[1] = *(const uint4*)&ar[kh * 16 + 8];
      const float* bp = Bb + (size_t)col * ldb + kb + kh * 16;
      #pragma unroll
      for (int e = 0; e < 16; ++e) bfr.s[e] = f2bf(bp[e]);
      c.v = __builtin_amdgcn_wmma_f32_16x16x32_bf16(false, a.v, false, bfr.v,
                                                    (short)0, c.v, false, false);
    }
  }
  if (!active) return;
  const int rbase = m0 + (kh << 3);
  #pragma unroll
  for (int v = 0; v < 8; ++v) {
    float x = c.f[v];
    size_t o = (size_t)cb + (size_t)(rbase + v) * ldc + col;
    if (outF) outF[o] = x;
    if (outB) outB[o] = f2bf(x);
  }
}

// ======================= weight packing =======================
// W[K][N] -> Wt[n*Kp + k] (pad with zeros)
__global__ void k_packw(const float* __restrict__ W, bf16_t* __restrict__ Wt,
                        int K, int N, int Kp, long long n) {
  long long i = (long long)blockIdx.x * blockDim.x + threadIdx.x;
  if (i >= n) return;
  int k = (int)(i % Kp);
  long long nn = i / Kp;
  float v = (k < K && nn < N) ? W[(size_t)k * N + nn] : 0.f;
  Wt[i] = f2bf(v);
}
// W already [N][K] row-major (conv weights) -> pad K only
__global__ void k_packw_rm(const float* __restrict__ Wnk, bf16_t* __restrict__ Wt,
                           int K, int Kp, long long n) {
  long long i = (long long)blockIdx.x * blockDim.x + threadIdx.x;
  if (i >= n) return;
  int k = (int)(i % Kp);
  long long nn = i / Kp;
  float v = (k < K) ? Wnk[(size_t)nn * K + k] : 0.f;
  Wt[i] = f2bf(v);
}

// ======================= layernorm =======================
__global__ __launch_bounds__(256) void k_ln(
    const float* __restrict__ x, const float* __restrict__ w, const float* __restrict__ bb,
    float* __restrict__ outF, bf16_t* __restrict__ outB, int Dm)
{
  const int row = blockIdx.x;
  const float* xr = x + (size_t)row * Dm;
  __shared__ float s1[256], s2[256];
  float a = 0.f, q = 0.f;
  for (int i = threadIdx.x; i < Dm; i += 256) { float v = xr[i]; a += v; q += v * v; }
  s1[threadIdx.x] = a; s2[threadIdx.x] = q;
  __syncthreads();
  for (int off = 128; off > 0; off >>= 1) {
    if (threadIdx.x < off) { s1[threadIdx.x] += s1[threadIdx.x + off]; s2[threadIdx.x] += s2[threadIdx.x + off]; }
    __syncthreads();
  }
  float mean = s1[0] / Dm;
  float var  = s2[0] / Dm - mean * mean;
  float inv  = rsqrtf(var + 1e-6f);
  for (int i = threadIdx.x; i < Dm; i += 256) {
    float v = (xr[i] - mean) * inv * w[i] + bb[i];
    size_t o = (size_t)row * Dm + i;
    if (outF) outF[o] = v;
    if (outB) outB[o] = f2bf(v);
  }
}

// ======================= residual + layer-scale =======================
__global__ void k_resid_ls(float* __restrict__ x, const float* __restrict__ y,
                           const float* __restrict__ ls, int Dm, long long n) {
  long long i = (long long)blockIdx.x * blockDim.x + threadIdx.x;
  if (i >= n) return;
  x[i] += ls[i % Dm] * y[i];
}

// ======================= softmax (masked, zero pad cols) =======================
__global__ __launch_bounds__(256) void k_softmax(float* __restrict__ S, int len, int valid) {
  float* row = S + (size_t)blockIdx.x * len;
  __shared__ float red[256];
  float m = -3.4e38f;
  for (int i = threadIdx.x; i < valid; i += 256) m = fmaxf(m, row[i]);
  red[threadIdx.x] = m; __syncthreads();
  for (int off = 128; off > 0; off >>= 1) {
    if (threadIdx.x < off) red[threadIdx.x] = fmaxf(red[threadIdx.x], red[threadIdx.x + off]);
    __syncthreads();
  }
  m = red[0]; __syncthreads();
  float s = 0.f;
  for (int i = threadIdx.x; i < valid; i += 256) {
    float e = __expf(row[i] - m); row[i] = e; s += e;
  }
  red[threadIdx.x] = s; __syncthreads();
  for (int off = 128; off > 0; off >>= 1) {
    if (threadIdx.x < off) red[threadIdx.x] += red[threadIdx.x + off];
    __syncthreads();
  }
  float inv = 1.0f / red[0];
  for (int i = threadIdx.x; i < valid; i += 256) row[i] *= inv;
  for (int i = valid + threadIdx.x; i < len; i += 256) row[i] = 0.f;
}

// ======================= V transpose-pack: qkv -> Vt[b][h][d][tokp] (zero pad) =======================
__global__ void k_pack_v(const float* __restrict__ qkv, float* __restrict__ vt,
                         int Ntokp, int valid, long long n) {
  long long i = (long long)blockIdx.x * blockDim.x + threadIdx.x;
  if (i >= n) return;
  int tok = (int)(i % Ntokp);
  long long r = i / Ntokp;
  int d = (int)(r % 64);
  long long bh = r / 64;
  int h = (int)(bh % 12), b = (int)(bh / 12);
  float v = 0.f;
  if (tok < valid)
    v = qkv[((size_t)b * Ntokp + tok) * 2304 + 1536 + h * 64 + d];
  vt[i] = v;
}

// ======================= antialiased bilinear resize (matches jax.image.resize) =======================
__global__ void k_resize_aa(const float* __restrict__ in, float* __restrict__ out,
                            int Hin, int Win, int Hout, int Wout) {
  int idx = blockIdx.x * blockDim.x + threadIdx.x;
  int ch = blockIdx.z;
  if (idx >= Hout * Wout) return;
  int oy = idx / Wout, ox = idx % Wout;
  const float* ip = in + (size_t)ch * Hin * Win;
  float sy = (float)Hin / Hout, sx = (float)Win / Wout;
  float cy = (oy + 0.5f) * sy - 0.5f;
  float cx = (ox + 0.5f) * sx - 0.5f;
  float ry = fmaxf(sy, 1.f), rx = fmaxf(sx, 1.f);
  int y0 = (int)ceilf(cy - ry);  if (y0 < 0) y0 = 0;
  int y1 = (int)floorf(cy + ry); if (y1 > Hin - 1) y1 = Hin - 1;
  int x0 = (int)ceilf(cx - rx);  if (x0 < 0) x0 = 0;
  int x1 = (int)floorf(cx + rx); if (x1 > Win - 1) x1 = Win - 1;
  float acc = 0.f, wsum = 0.f;
  for (int y = y0; y <= y1; ++y) {
    float wy = 1.f - fabsf((y - cy) / ry); if (wy <= 0.f) continue;
    for (int x = x0; x <= x1; ++x) {
      float wx = 1.f - fabsf((x - cx) / rx); if (wx <= 0.f) continue;
      float w = wy * wx;
      acc += w * ip[(size_t)y * Win + x];
      wsum += w;
    }
  }
  out[(size_t)ch * Hout * Wout + idx] = acc / wsum;
}

// ======================= im2col for 14x14 stride-14 patch conv (bf16, K padded to 640) =======================
__global__ void k_im2col(const float* __restrict__ img, bf16_t* __restrict__ out,
                         int Himg, int Gdim, int rowsP, long long n) {
  long long i = (long long)blockIdx.x * blockDim.x + threadIdx.x;
  if (i >= n) return;
  int k = (int)(i % 640);
  long long rr = i / 640;
  int r = (int)(rr % rowsP);
  int b = (int)(rr / rowsP);
  float v = 0.f;
  if (r < Gdim * Gdim && k < 588) {
    int gy = r / Gdim, gx = r % Gdim;
    int c = k / 196, rem = k % 196, py = rem / 14, px = rem % 14;
    v = img[(((size_t)b * 3 + c) * Himg + gy * 14 + py) * Himg + gx * 14 + px];
  }
  out[i] = f2bf(v);
}

// ======================= token assembly =======================
__global__ void k_assemble_sel(const float* __restrict__ t, const float* __restrict__ pos,
                               const float* __restrict__ cls, const float* __restrict__ reg,
                               float* __restrict__ x, long long n) {
  long long i = (long long)blockIdx.x * blockDim.x + threadIdx.x;
  if (i >= n) return;
  int c = (int)(i % 768);
  long long rr = i / 768;
  int tok = (int)(rr % 128);
  int b = (int)(rr / 128);
  float v = 0.f;
  if (tok == 0) v = cls[c];
  else if (tok < 5) v = reg[(tok - 1) * 768 + c];
  else if (tok < 126)
    v = t[((size_t)b * 128 + (tok - 5)) * 768 + c] + pos[(size_t)(tok - 5) * 768 + c];
  x[i] = v;
}

__global__ void k_assemble_ext(const float* __restrict__ selln, const float* __restrict__ te,
                               const float* __restrict__ pos, const int* __restrict__ idx,
                               float* __restrict__ x, int Ntokp, long long n) {
  long long i = (long long)blockIdx.x * blockDim.x + threadIdx.x;
  if (i >= n) return;
  int c = (int)(i % 768);
  long long rr = i / 768;
  int tok = (int)(rr % Ntokp);
  int b = (int)(rr / Ntokp);
  float v = 0.f;
  if (tok < 5) v = selln[((size_t)b * 128 + tok) * 768 + c];
  else if (tok < 517) {
    int j = idx[b * 512 + (tok - 5)];
    v = te[((size_t)b * 1376 + j) * 768 + c] + pos[(size_t)j * 768 + c];
  }
  x[i] = v;
}

// ======================= saliency map reshuffle + top-k =======================
__global__ void k_smap44(const float* __restrict__ smap2, float* __restrict__ full, long long n) {
  long long i = (long long)blockIdx.x * blockDim.x + threadIdx.x;
  if (i >= n) return;
  int xx = (int)(i % 44);
  long long rr = i / 44;
  int yy = (int)(rr % 44);
  int b = (int)(rr / 44);
  int gy = yy / 4, ry = yy % 4, gx = xx / 4, rx = xx % 4;
  full[i] = smap2[((size_t)b * 128 + 5 + gy * 11 + gx) * 16 + ry * 4 + rx];
}

__global__ __launch_bounds__(512) void k_topk(const float* __restrict__ up, int* __restrict__ outIdx) {
  __shared__ float v[2048];
  __shared__ int   id[2048];
  const int b = blockIdx.x;
  for (int i = threadIdx.x; i < 2048; i += 512) {
    if (i < 1369) { v[i] = up[b * 1369 + i]; id[i] = i; }
    else          { v[i] = -3.4e38f;         id[i] = 1 << 30; }
  }
  __syncthreads();
  for (int k = 2; k <= 2048; k <<= 1) {
    for (int j = k >> 1; j > 0; j >>= 1) {
      for (int i = threadIdx.x; i < 2048; i += 512) {
        int ixj = i ^ j;
        if (ixj > i) {
          bool dir = ((i & k) == 0);  // descending runs
          float va = v[i], vb = v[ixj];
          int ia = id[i], ib = id[ixj];
          bool aFirst = (va > vb) || (va == vb && ia < ib);
          bool sw = dir ? !aFirst : aFirst;
          if (sw) { v[i] = vb; v[ixj] = va; id[i] = ib; id[ixj] = ia; }
        }
      }
      __syncthreads();
    }
  }
  for (int i = threadIdx.x; i < 512; i += 512) outIdx[b * 512 + i] = id[i];
}

// ======================= misc small kernels =======================
__global__ void k_cls(const bf16_t* __restrict__ lnbf, bf16_t* __restrict__ clsb, int Ntokp, int n) {
  int i = blockIdx.x * blockDim.x + threadIdx.x;
  if (i >= n) return;
  int c = i % 768, r = i / 768;
  clsb[i] = (r < 4) ? lnbf[((size_t)r * Ntokp) * 768 + c] : (bf16_t)0;
}
__global__ void k_out(const float* __restrict__ tmp, const float* __restrict__ hb,
                      float* __restrict__ out, int n) {
  int i = blockIdx.x * blockDim.x + threadIdx.x;
  if (i >= n) return;
  int b = i / 1000, c = i % 1000;
  out[i] = tmp[(size_t)b * 1008 + c] + hb[c];
}

// ======================= host-side transformer-stack driver =======================
static void run_blocks(hipStream_t st, float* x, int Bn, int Ntokp, int valid, int L,
                       const bf16_t* qkvW, const bf16_t* projW, const bf16_t* fc1W, const bf16_t* fc2W,
                       const float* ln1w, const float* ln1b, const float* qkvb,
                       const float* projb, const float* ls1,
                       const float* ln2w, const float* ln2b,
                       const float* fc1b, const float* fc2b, const float* ls2,
                       bf16_t* lnbf, float* qkv, float* S, float* Vt,
                       bf16_t* attnO, float* ybuf, bf16_t* hid)
{
  const long long QS = 2304LL * 768, PS = 768LL * 768, F1S = 3072LL * 768, F2S = 768LL * 3072;
  const int M = Bn * Ntokp;
  const long long NT2 = (long long)Ntokp * Ntokp;
  for (int l = 0; l < L; ++l) {
    k_ln<<<M, 256, 0, st>>>(x, ln1w + l * 768, ln1b + l * 768, nullptr, lnbf, 768);
    k_gemm_wp<<<dim3(M / 32, (2304 / 16 + 7) / 8), 256, 0, st>>>(
        lnbf, 768, qkvW + l * QS, 768, qkvb + (long long)l * 2304,
        qkv, nullptr, 2304, 2304, 768, 0);
    // scores: S[b,h,:,:] = (Q*scale) @ K^T
    k_bgemm_bt<<<dim3(Ntokp / 16, (Ntokp / 16 + 7) / 8, Bn * 12), 256, 0, st>>>(
        qkv, (long long)Ntokp * 2304, 64, 2304,
        qkv + 768, (long long)Ntokp * 2304, 64, 2304,
        S, nullptr, 12LL * NT2, NT2, Ntokp,
        Ntokp, 64, 12, 0.125f);
    k_softmax<<<Bn * 12 * Ntokp, 256, 0, st>>>(S, Ntokp, valid);
    {
      long long n = (long long)Bn * 12 * 64 * Ntokp;
      k_pack_v<<<(unsigned)((n + 255) / 256), 256, 0, st>>>(qkv, Vt, Ntokp, valid, n);
    }
    // attn @ V -> attnO (bf16, interleaved heads into [tok,768])
    k_bgemm_bt<<<dim3(Ntokp / 16, 1, Bn * 12), 256, 0, st>>>(
        S, 12LL * NT2, NT2, Ntokp,
        Vt, 12LL * 64 * Ntokp, 64LL * Ntokp, Ntokp,
        nullptr, attnO, (long long)Ntokp * 768, 64, 768,
        64, Ntokp, 12, 1.0f);
    k_gemm_wp<<<dim3(M / 32, 6), 256, 0, st>>>(
        attnO, 768, projW + l * PS, 768, projb + (long long)l * 768,
        ybuf, nullptr, 768, 768, 768, 0);
    {
      long long n = (long long)M * 768;
      k_resid_ls<<<(unsigned)((n + 255) / 256), 256, 0, st>>>(x, ybuf, ls1 + l * 768, 768, n);
    }
    k_ln<<<M, 256, 0, st>>>(x, ln2w + l * 768, ln2b + l * 768, nullptr, lnbf, 768);
    k_gemm_wp<<<dim3(M / 32, 24), 256, 0, st>>>(
        lnbf, 768, fc1W + l * F1S, 768, fc1b + (long long)l * 3072,
        nullptr, hid, 3072, 3072, 768, 1 /*gelu*/);
    k_gemm_wp<<<dim3(M / 32, 6), 256, 0, st>>>(
        hid, 3072, fc2W + l * F2S, 3072, fc2b + (long long)l * 768,
        ybuf, nullptr, 768, 768, 3072, 0);
    {
      long long n = (long long)M * 768;
      k_resid_ls<<<(unsigned)((n + 255) / 256), 256, 0, st>>>(x, ybuf, ls2 + l * 768, 768, n);
    }
  }
}

// ======================= launch =======================
extern "C" void kernel_launch(void* const* d_in, const int* in_sizes, int n_in,
                              void* d_out, int out_size, void* d_ws, size_t ws_size,
                              hipStream_t stream) {
  (void)in_sizes; (void)n_in; (void)out_size; (void)ws_size;
  // ---- input pointers (setup_inputs flattening order; _blocks has 14 entries incl. ls2) ----
  const float* images  = (const float*)d_in[0];
  const float* s_pw    = (const float*)d_in[1];
  const float* s_pb    = (const float*)d_in[2];
  const float* s_pos   = (const float*)d_in[3];
  const float* s_cls   = (const float*)d_in[4];
  const float* s_reg   = (const float*)d_in[5];
  const float* s_ln1w  = (const float*)d_in[6];
  const float* s_ln1b  = (const float*)d_in[7];
  const float* s_qkvw  = (const float*)d_in[8];
  const float* s_qkvb  = (const float*)d_in[9];
  const float* s_projw = (const float*)d_in[10];
  const float* s_projb = (const float*)d_in[11];
  const float* s_ls1   = (const float*)d_in[12];
  const float* s_ln2w  = (const float*)d_in[13];
  const float* s_ln2b  = (const float*)d_in[14];
  const float* s_fc1w  = (const float*)d_in[15];
  const float* s_fc1b  = (const float*)d_in[16];
  const float* s_fc2w  = (const float*)d_in[17];
  const float* s_fc2b  = (const float*)d_in[18];
  const float* s_ls2   = (const float*)d_in[19];
  const float* s_nw    = (const float*)d_in[20];
  const float* s_nb    = (const float*)d_in[21];
  const float* s_hw1   = (const float*)d_in[22];
  const float* s_hb1   = (const float*)d_in[23];
  const float* s_hw2   = (const float*)d_in[24];
  const float* s_hb2   = (const float*)d_in[25];
  const float* e_pw    = (const float*)d_in[26];
  const float* e_pb    = (const float*)d_in[27];
  const float* e_pos   = (const float*)d_in[28];
  const float* e_ln1w  = (const float*)d_in[29];
  const float* e_ln1b  = (const float*)d_in[30];
  const float* e_qkvw  = (const float*)d_in[31];
  const float* e_qkvb  = (const float*)d_in[32];
  const float* e_projw = (const float*)d_in[33];
  const float* e_projb = (const float*)d_in[34];
  const float* e_ls1   = (const float*)d_in[35];
  const float* e_ln2w  = (const float*)d_in[36];
  const float* e_ln2b  = (const float*)d_in[37];
  const float* e_fc1w  = (const float*)d_in[38];
  const float* e_fc1b  = (const float*)d_in[39];
  const float* e_fc2w  = (const float*)d_in[40];
  const float* e_fc2b  = (const float*)d_in[41];
  const float* e_ls2   = (const float*)d_in[42];
  const float* e_nw    = (const float*)d_in[43];
  const float* e_nb    = (const float*)d_in[44];
  const float* head_w  = (const float*)d_in[45];
  const float* head_b  = (const float*)d_in[46];

  // ---- dims ----
  const int NSP = 128, NSV = 126, NEP = 544, NEV = 517;
  const int MS = 4 * NSP, ME = 4 * NEP;
  const int KC = 640;  // padded im2col K (588 -> 640, multiple of 64)
  const long long QS = 2304LL * 768, PS = 768LL * 768, F1S = 3072LL * 768, F2S = 768LL * 3072;

  // ---- workspace carve ----
  char* p = (char*)d_ws;
  auto alloc = [&](size_t elems, size_t esz) -> char* {
    char* r = p;
    p += (elems * esz + 255) & ~(size_t)255;
    return r;
  };
  bf16_t* selPatchWt = (bf16_t*)alloc(768LL * KC, 2);
  bf16_t* extPatchWt = (bf16_t*)alloc(768LL * KC, 2);
  bf16_t* selQkvWt   = (bf16_t*)alloc(3 * QS, 2);
  bf16_t* selProjWt  = (bf16_t*)alloc(3 * PS, 2);
  bf16_t* selFc1Wt   = (bf16_t*)alloc(3 * F1S, 2);
  bf16_t* selFc2Wt   = (bf16_t*)alloc(3 * F2S, 2);
  bf16_t* extQkvWt   = (bf16_t*)alloc(12 * QS, 2);
  bf16_t* extProjWt  = (bf16_t*)alloc(12 * PS, 2);
  bf16_t* extFc1Wt   = (bf16_t*)alloc(12 * F1S, 2);
  bf16_t* extFc2Wt   = (bf16_t*)alloc(12 * F2S, 2);
  bf16_t* h1Wt       = (bf16_t*)alloc(3072LL * 768, 2);
  bf16_t* h2Wt       = (bf16_t*)alloc(16LL * 3072, 2);
  bf16_t* hfWt       = (bf16_t*)alloc(1008LL * 768, 2);

  float*  imgLR   = (float*)alloc(4LL * 3 * 154 * 154, 4);
  bf16_t* colSel  = (bf16_t*)alloc((long long)MS * KC, 2);
  float*  tSel    = (float*)alloc((long long)MS * 768, 4);
  float*  xSel    = (float*)alloc((long long)MS * 768, 4);
  float*  selLnF  = (float*)alloc((long long)MS * 768, 4);
  bf16_t* lnbf    = (bf16_t*)alloc((long long)ME * 768, 2);
  float*  qkv     = (float*)alloc((long long)ME * 2304, 4);
  float*  S       = (float*)alloc(4LL * 12 * NEP * NEP, 4);
  float*  Vt      = (float*)alloc(4LL * 12 * 64 * NEP, 4);
  bf16_t* attnO   = (bf16_t*)alloc((long long)ME * 768, 2);
  float*  ybuf    = (float*)alloc((long long)ME * 768, 4);
  bf16_t* hid     = (bf16_t*)alloc((long long)ME * 3072, 2);
  float*  smap2   = (float*)alloc((long long)MS * 16, 4);
  float*  full44  = (float*)alloc(4LL * 44 * 44, 4);
  float*  up37    = (float*)alloc(4LL * 1369, 4);
  int*    topIdx  = (int*)alloc(4LL * 512, 4);
  bf16_t* colExt  = (bf16_t*)alloc(4LL * 1376 * KC, 2);
  float*  te      = (float*)alloc(4LL * 1376 * 768, 4);
  float*  xExt    = (float*)alloc((long long)ME * 768, 4);
  bf16_t* clsb    = (bf16_t*)alloc(32LL * 768, 2);
  float*  headTmp = (float*)alloc(32LL * 1008, 4);

  auto packT = [&](const float* W, bf16_t* Wt, int K, int N, int Kp, int Np) {
    long long n = (long long)Np * Kp;
    k_packw<<<(unsigned)((n + 255) / 256), 256, 0, stream>>>(W, Wt, K, N, Kp, n);
  };

  // ---- weight packing (fp32 -> K-major bf16) ----
  for (int l = 0; l < 3; ++l) {
    packT(s_qkvw + (long long)l * 768 * 2304, selQkvWt + l * QS, 768, 2304, 768, 2304);
    packT(s_projw + (long long)l * 768 * 768, selProjWt + l * PS, 768, 768, 768, 768);
    packT(s_fc1w + (long long)l * 768 * 3072, selFc1Wt + l * F1S, 768, 3072, 768, 3072);
    packT(s_fc2w + (long long)l * 3072 * 768, selFc2Wt + l * F2S, 3072, 768, 3072, 768);
  }
  for (int l = 0; l < 12; ++l) {
    packT(e_qkvw + (long long)l * 768 * 2304, extQkvWt + l * QS, 768, 2304, 768, 2304);
    packT(e_projw + (long long)l * 768 * 768, extProjWt + l * PS, 768, 768, 768, 768);
    packT(e_fc1w + (long long)l * 768 * 3072, extFc1Wt + l * F1S, 768, 3072, 768, 3072);
    packT(e_fc2w + (long long)l * 3072 * 768, extFc2Wt + l * F2S, 3072, 768, 3072, 768);
  }
  packT(s_hw1, h1Wt, 768, 3072, 768, 3072);
  packT(s_hw2, h2Wt, 3072, 16, 3072, 16);
  packT(head_w, hfWt, 768, 1000, 768, 1008);
  {
    long long n = 768LL * KC;
    k_packw_rm<<<(unsigned)((n + 255) / 256), 256, 0, stream>>>(s_pw, selPatchWt, 588, KC, n);
    k_packw_rm<<<(unsigned)((n + 255) / 256), 256, 0, stream>>>(e_pw, extPatchWt, 588, KC, n);
  }

  // ---- selector path ----
  k_resize_aa<<<dim3((154 * 154 + 255) / 256, 1, 12), 256, 0, stream>>>(
      images, imgLR, 518, 518, 154, 154);
  {
    long long n = (long long)MS * KC;
    k_im2col<<<(unsigned)((n + 255) / 256), 256, 0, stream>>>(imgLR, colSel, 154, 11, 128, n);
  }
  k_gemm_wp<<<dim3(MS / 32, 6), 256, 0, stream>>>(
      colSel, KC, selPatchWt, KC, s_pb, tSel, nullptr, 768, 768, KC, 0);
  {
    long long n = (long long)MS * 768;
    k_assemble_sel<<<(unsigned)((n + 255) / 256), 256, 0, stream>>>(tSel, s_pos, s_cls, s_reg, xSel, n);
  }
  run_blocks(stream, xSel, 4, NSP, NSV, 3,
             selQkvWt, selProjWt, selFc1Wt, selFc2Wt,
             s_ln1w, s_ln1b, s_qkvb, s_projb, s_ls1,
             s_ln2w, s_ln2b, s_fc1b, s_fc2b, s_ls2,
             lnbf, qkv, S, Vt, attnO, ybuf, hid);
  k_ln<<<MS, 256, 0, stream>>>(xSel, s_nw, s_nb, selLnF, lnbf, 768);

  // saliency head: gelu(pt @ w1 + b1) @ w2 + b2  (run over all padded rows; consumers pick token rows)
  k_gemm_wp<<<dim3(MS / 32, 24), 256, 0, stream>>>(
      lnbf, 768, h1Wt, 768, s_hb1, nullptr, hid, 3072, 3072, 768, 1);
  k_gemm_wp<<<dim3(MS / 32, 1), 256, 0, stream>>>(
      hid, 3072, h2Wt, 3072, s_hb2, smap2, nullptr, 16, 16, 3072, 0);
  {
    long long n = 4LL * 44 * 44;
    k_smap44<<<(unsigned)((n + 255) / 256), 256, 0, stream>>>(smap2, full44, n);
  }
  k_resize_aa<<<dim3((37 * 37 + 255) / 256, 1, 4), 256, 0, stream>>>(
      full44, up37, 44, 44, 37, 37);
  k_topk<<<4, 512, 0, stream>>>(up37, topIdx);

  // ---- extractor path ----
  {
    long long n = 4LL * 1376 * KC;
    k_im2col<<<(unsigned)((n + 255) / 256), 256, 0, stream>>>(images, colExt, 518, 37, 1376, n);
  }
  k_gemm_wp<<<dim3(4 * 1376 / 32, 6), 256, 0, stream>>>(
      colExt, KC, extPatchWt, KC, e_pb, te, nullptr, 768, 768, KC, 0);
  {
    long long n = (long long)ME * 768;
    k_assemble_ext<<<(unsigned)((n + 255) / 256), 256, 0, stream>>>(
        selLnF, te, e_pos, topIdx, xExt, NEP, n);
  }
  run_blocks(stream, xExt, 4, NEP, NEV, 12,
             extQkvWt, extProjWt, extFc1Wt, extFc2Wt,
             e_ln1w, e_ln1b, e_qkvb, e_projb, e_ls1,
             e_ln2w, e_ln2b, e_fc1b, e_fc2b, e_ls2,
             lnbf, qkv, S, Vt, attnO, ybuf, hid);
  k_ln<<<ME, 256, 0, stream>>>(xExt, e_nw, e_nb, nullptr, lnbf, 768);
  k_cls<<<(32 * 768 + 255) / 256, 256, 0, stream>>>(lnbf, clsb, NEP, 32 * 768);
  k_gemm_wp<<<dim3(1, 8), 256, 0, stream>>>(
      clsb, 768, hfWt, 768, nullptr, headTmp, nullptr, 1008, 1008, 768, 0);
  k_out<<<16, 256, 0, stream>>>(headTmp, head_b, (float*)d_out, 4000);
}
